// LengthRegulator_42314017800606
// MI455X (gfx1250) — matrix operand order repaired and checked by reading, not compiled, not run on hardware
//
#include <hip/hip_runtime.h>
#include <hip/hip_bf16.h>

// ---------------------------------------------------------------------------
// FastSpeech-style length regulator + duration predictor for MI455X (gfx1250)
//   conv1d(K=3) as GEMM  M=32768, N=256, Kdep=768 via v_wmma_f32_16x16x32_bf16
//   Software-pipelined: B-fragment ds_loads double-buffered against WMMA,
//   next-chunk global staging loads issued under current-chunk compute.
//   LayerNorm fused in epilogue (wave-local half-wave shfl reductions)
//   length-regulate gather: binary search over per-batch cumsum, NT stores
// ---------------------------------------------------------------------------

typedef __attribute__((ext_vector_type(16))) __bf16       v16bf;
typedef __attribute__((ext_vector_type(8)))  float        v8f;
typedef __attribute__((ext_vector_type(4)))  float        f4;
typedef __attribute__((ext_vector_type(4)))  unsigned int u4;

#define B_      32
#define L_      1024
#define E_      256
#define H_      256
#define TMEL_   8192
#define NCHUNK  24      // Kdep 768 / 32

union Frag { v16bf v; u4 q[2]; };

// --- pack conv weights w[h][e][k] (fp32) -> packB[c][h][j] (bf16), kk=k*256+e
__global__ void pack_w_kernel(const float* __restrict__ w,
                              __bf16* __restrict__ packB) {
  int idx = blockIdx.x * 256 + threadIdx.x;     // 0 .. 768*256-1
  int j = idx & 31;
  int h = (idx >> 5) & 255;
  int c = idx >> 13;
  int kk = c * 32 + j;
  int k = kk >> 8;
  int e = kk & 255;
  packB[idx] = (__bf16)w[(h * 256 + e) * 3 + k];
}

// --- A fragment 16x32 bf16, ISA lane layout, zero-fill conv padding -------
template <bool IN_BF16>
__device__ __forceinline__ void load_afrag(const void* __restrict__ in_v,
                                           int bidx, int lpos, int half,
                                           int c, Frag& af) {
  const int k    = c >> 3;
  const int e0   = (c & 7) * 32;
  const int srcl = lpos + k - 1;
  if (srcl >= 0 && srcl < L_) {
    const size_t rb = ((size_t)(bidx * L_ + srcl)) * E_ + e0 + half * 8;
    if (IN_BF16) {
      const __bf16* rp = (const __bf16*)in_v + rb;
      af.q[0] = *(const u4*)(rp);
      af.q[1] = *(const u4*)(rp + 16);
    } else {
      const float* rp = (const float*)in_v + rb;
      const f4 f0 = *(const f4*)(rp);
      const f4 f1 = *(const f4*)(rp + 4);
      const f4 f2 = *(const f4*)(rp + 16);
      const f4 f3 = *(const f4*)(rp + 20);
#pragma unroll
      for (int i = 0; i < 4; ++i) {
        af.v[i]      = (__bf16)f0[i];
        af.v[4 + i]  = (__bf16)f1[i];
        af.v[8 + i]  = (__bf16)f2[i];
        af.v[12 + i] = (__bf16)f3[i];
      }
    }
  } else {
#pragma unroll
    for (int i = 0; i < 16; ++i) af.v[i] = (__bf16)0.f;
  }
}

// --- fused conv1d(K=3, pad=1) + bias + LayerNorm + ReLU, bf16 out ---------
template <bool IN_BF16>
__global__ __launch_bounds__(256) void conv_ln_relu_kernel(
    const void* __restrict__ in_v, const __bf16* __restrict__ packB,
    const float* __restrict__ cb, const float* __restrict__ gamma,
    const float* __restrict__ beta, __bf16* __restrict__ out) {
  __shared__ __align__(16) unsigned short ldsB[256 * 32];  // 16 KB bf16 chunk

  const int tid    = threadIdx.x;
  const int wave   = tid >> 5;
  const int lane   = tid & 31;
  const int lane15 = lane & 15;
  const int half   = lane >> 4;

  const int rowA = blockIdx.x * 128 + wave * 16;  // 128 rows/block, in-batch
  const int rowG = rowA + lane15;                 // A-matrix row of this lane
  const int bidx = rowG >> 10;
  const int lpos = rowG & 1023;

  v8f acc[16];
#pragma unroll
  for (int t = 0; t < 16; ++t)
#pragma unroll
    for (int i = 0; i < 8; ++i) acc[t][i] = 0.f;

  // prologue: chunk-0 B staging data + chunk-0 A fragment already in flight
  u4 breg[4];
  {
    const u4* src = (const u4*)(packB);
#pragma unroll
    for (int i = 0; i < 4; ++i) breg[i] = src[tid * 4 + i];
  }
  Frag af;
  load_afrag<IN_BF16>(in_v, bidx, lpos, half, 0, af);

#pragma unroll 1
  for (int c = 0; c < NCHUNK; ++c) {
    __syncthreads();  // all waves done reading previous chunk's LDS
    {
      u4* dst = (u4*)ldsB;
#pragma unroll
      for (int i = 0; i < 4; ++i) dst[tid * 4 + i] = breg[i];
    }
    // prefetch next chunk's global data under this chunk's compute
    if (c + 1 < NCHUNK) {
      const u4* src = (const u4*)(packB + (size_t)(c + 1) * 256 * 32);
#pragma unroll
      for (int i = 0; i < 4; ++i) breg[i] = src[tid * 4 + i];
    }
    Frag afc = af;
    if (c + 1 < NCHUNK)
      load_afrag<IN_BF16>(in_v, bidx, lpos, half, c + 1, af);
    __syncthreads();  // staged chunk visible

    // ---- 16 N-tiles, B fragments double-buffered against WMMA -----------
    Frag bf0, bf1;
    {
      const u4* ls = (const u4*)&ldsB[lane15 * 32 + half * 8];
      bf0.q[0] = ls[0];
      bf0.q[1] = ls[2];
    }
#pragma unroll
    for (int t = 0; t < 16; ++t) {
      Frag& cur = (t & 1) ? bf1 : bf0;
      Frag& nxt = (t & 1) ? bf0 : bf1;
      if (t < 15) {
        const u4* ls = (const u4*)&ldsB[(16 * (t + 1) + lane15) * 32 + half * 8];
        nxt.q[0] = ls[0];
        nxt.q[1] = ls[2];
      }
      acc[t] = __builtin_amdgcn_wmma_f32_16x16x32_bf16(
          false, afc.v, false, cur.v, (short)0, acc[t], false, false);
    }
  }

  // ---- epilogue: +bias, LayerNorm over 256 cols (wave-local), affine+ReLU
#pragma unroll
  for (int t = 0; t < 16; ++t) {
    const float cbv = cb[16 * t + lane15];
#pragma unroll
    for (int j = 0; j < 8; ++j) acc[t][j] += cbv;
  }
  float s[8], ss[8];
#pragma unroll
  for (int j = 0; j < 8; ++j) {
    float a = 0.f, b = 0.f;
#pragma unroll
    for (int t = 0; t < 16; ++t) {
      const float v = acc[t][j];
      a += v;
      b += v * v;
    }
    s[j] = a;
    ss[j] = b;
  }
#pragma unroll
  for (int m = 1; m < 16; m <<= 1) {
#pragma unroll
    for (int j = 0; j < 8; ++j) {
      s[j]  += __shfl_xor(s[j], m, 32);
      ss[j] += __shfl_xor(ss[j], m, 32);
    }
  }
  float mean[8], rstd[8];
#pragma unroll
  for (int j = 0; j < 8; ++j) {
    mean[j] = s[j] * (1.f / 256.f);
    const float var = ss[j] * (1.f / 256.f) - mean[j] * mean[j];
    rstd[j] = rsqrtf(var + 1e-5f);
  }
#pragma unroll
  for (int t = 0; t < 16; ++t) {
    const int col = 16 * t + lane15;
    const float gv = gamma[col], bv = beta[col];
#pragma unroll
    for (int j = 0; j < 8; ++j) {
      const int row = rowA + half * 8 + j;  // C/D layout: vgpr j, half-wave
      float v = (acc[t][j] - mean[j]) * rstd[j] * gv + bv;
      v = fmaxf(v, 0.f);
      out[(size_t)row * H_ + col] = (__bf16)v;
    }
  }
}

// --- final linear H->1 + ReLU: one wave per row ---------------------------
__global__ __launch_bounds__(256) void linear_relu_kernel(
    const __bf16* __restrict__ y, const float* __restrict__ lw,
    const float* __restrict__ lb, float* __restrict__ dur) {
  const int row  = (int)((blockIdx.x * 256 + threadIdx.x) >> 5);
  const int lane = threadIdx.x & 31;
  const __bf16* rp = y + (size_t)row * H_ + lane * 8;
  float s = 0.f;
#pragma unroll
  for (int i = 0; i < 8; ++i) s += (float)rp[i] * lw[lane * 8 + i];
#pragma unroll
  for (int m = 16; m >= 1; m >>= 1) s += __shfl_xor(s, m, 32);
  if (lane == 0) dur[row] = fmaxf(s + lb[0], 0.f);
}

// --- inclusive cumsum of target[b][0..1023] (Hillis-Steele in LDS) --------
__global__ __launch_bounds__(1024) void cumsum_kernel(
    const int* __restrict__ target, int* __restrict__ cum) {
  __shared__ int sh[1024];
  const int b = blockIdx.x, t = threadIdx.x;
  sh[t] = target[b * 1024 + t];
  __syncthreads();
  for (int off = 1; off < 1024; off <<= 1) {
    const int v = (t >= off) ? sh[t - off] : 0;
    __syncthreads();
    sh[t] += v;
    __syncthreads();
  }
  cum[b * 1024 + t] = sh[t];
}

// --- length regulate: 64 mel rows per block, NT streaming stores ----------
__global__ __launch_bounds__(256) void gather_kernel(
    const float* __restrict__ x, const int* __restrict__ cum,
    float* __restrict__ out) {
  __shared__ int shc[1024];
  __shared__ int shidx[64];
  __shared__ int shval[64];
  const int b   = blockIdx.x >> 7;           // 128 blocks per batch
  const int t0  = (blockIdx.x & 127) * 64;
  const int tid = threadIdx.x;
  for (int i = tid; i < 1024; i += 256) shc[i] = cum[b * 1024 + i];
  __syncthreads();
  const int total = shc[1023];
  if (tid < 64) {
    const int t = t0 + tid;
    int lo = 0, hi = 1024;                   // searchsorted(cum, t, 'right')
    while (lo < hi) {
      const int mid = (lo + hi) >> 1;
      if (shc[mid] <= t) lo = mid + 1; else hi = mid;
    }
    shidx[tid] = lo < 1023 ? lo : 1023;
    shval[tid] = (t < total) ? 1 : 0;
  }
  __syncthreads();
  const f4* xb = (const f4*)(x + (size_t)b * L_ * E_);
  f4* ob = (f4*)(out + ((size_t)b * TMEL_ + t0) * E_);
#pragma unroll
  for (int i = 0; i < 16; ++i) {
    const int linear = tid + 256 * i;        // 4096 f4 slots = 64 rows x 64
    const int r  = linear >> 6;
    const int c4 = linear & 63;
    f4 v;
    if (shval[r]) {
      v = xb[shidx[r] * 64 + c4];
    } else {
      v = (f4){0.f, 0.f, 0.f, 0.f};
    }
    __builtin_nontemporal_store(v, ob + r * 64 + c4);
  }
}

// ---------------------------------------------------------------------------
extern "C" void kernel_launch(void* const* d_in, const int* in_sizes, int n_in,
                              void* d_out, int out_size, void* d_ws,
                              size_t ws_size, hipStream_t stream) {
  const float* x      = (const float*)d_in[0];
  const int*   target = (const int*)d_in[1];
  // d_in[2] = mel_max_len (fixed 8192)
  const float* c1w = (const float*)d_in[3];
  const float* c1b = (const float*)d_in[4];
  const float* g1  = (const float*)d_in[5];
  const float* b1  = (const float*)d_in[6];
  const float* c2w = (const float*)d_in[7];
  const float* c2b = (const float*)d_in[8];
  const float* g2  = (const float*)d_in[9];
  const float* b2  = (const float*)d_in[10];
  const float* lw  = (const float*)d_in[11];
  const float* lb  = (const float*)d_in[12];

  float* out = (float*)d_out;                       // [32, 8192, 256]
  float* dur = out + (size_t)B_ * TMEL_ * E_;       // [32, 1024, 1]

  // workspace layout
  char* ws = (char*)d_ws;
  __bf16* packB1 = (__bf16*)ws;                               // 384 KB
  __bf16* packB2 = packB1 + 768 * 256;                        // 384 KB
  __bf16* y1 = (__bf16*)(ws + 2u * 768 * 256 * sizeof(__bf16));  // 16 MB
  __bf16* y2 = y1 + (size_t)B_ * L_ * H_;                     // 16 MB
  int* cum = (int*)((char*)(y2 + (size_t)B_ * L_ * H_));      // 128 KB

  // 1. pack conv weights into WMMA-B bf16 layout
  pack_w_kernel<<<768, 256, 0, stream>>>(c1w, packB1);
  pack_w_kernel<<<768, 256, 0, stream>>>(c2w, packB2);

  // 2. duration predictor
  conv_ln_relu_kernel<false><<<256, 256, 0, stream>>>(x, packB1, c1b, g1, b1, y1);
  conv_ln_relu_kernel<true><<<256, 256, 0, stream>>>(y1, packB2, c2b, g2, b2, y2);
  linear_relu_kernel<<<4096, 256, 0, stream>>>(y2, lw, lb, dur);

  // 3. length regulator (uses `target` durations)
  cumsum_kernel<<<B_, 1024, 0, stream>>>(target, cum);
  gather_kernel<<<B_ * (TMEL_ / 64), 256, 0, stream>>>(x, cum, out);
}
